// SparseResNet2D_13125420057181
// MI455X (gfx1250) — matrix operand order repaired and checked by reading, not compiled, not run on hardware
//
#include <hip/hip_runtime.h>

typedef __attribute__((ext_vector_type(16))) _Float16 v16h;
typedef __attribute__((ext_vector_type(8)))  float    v8f;

#define EPS_BN 1e-4f

// ---------------------------------------------------------------- utilities
static __global__ void k_zero(float* __restrict__ p, int n) {
  int i = blockIdx.x * blockDim.x + threadIdx.x;
  if (i < n) p[i] = 0.f;
}

// ------------------------------------------------------------------ scatter
static __global__ void k_scatter(const int* __restrict__ coords, const float* __restrict__ feats,
                                 float* __restrict__ dense, float* __restrict__ mask, int n) {
  int i = blockIdx.x * blockDim.x + threadIdx.x;
  if (i >= n) return;
  int y = coords[3 * i + 0];
  int x = coords[3 * i + 1];
  int b = coords[3 * i + 2];
  int p = (b * 255 + y) * 255 + x;
  atomicAdd(&dense[2 * p + 0], feats[2 * i + 0]);
  atomicAdd(&dense[2 * p + 1], feats[2 * i + 1]);
  mask[p] = 1.f;
}

// ----------------------------------------------------- stem conv (2->16), SAME
static __global__ void k_stem(const float* __restrict__ dense, const float* __restrict__ mask,
                              const float* __restrict__ w, _Float16* __restrict__ out) {
  const int HW = 255 * 255;
  int p = blockIdx.x * blockDim.x + threadIdx.x;
  if (p >= 32 * HW) return;
  int b = p / HW;
  int r = p - b * HW;
  int y = r / 255, x = r - (r / 255) * 255;
  float acc[16];
#pragma unroll
  for (int c = 0; c < 16; ++c) acc[c] = 0.f;
  for (int ky = 0; ky < 3; ++ky) {
    int iy = y - 1 + ky;
    if (iy < 0 || iy >= 255) continue;
    for (int kx = 0; kx < 3; ++kx) {
      int ix = x - 1 + kx;
      if (ix < 0 || ix >= 255) continue;
      const float* px = dense + ((b * 255 + iy) * 255 + ix) * 2;
      const float* wt = w + (ky * 3 + kx) * 2 * 16;
      float f0 = px[0], f1 = px[1];
#pragma unroll
      for (int c = 0; c < 16; ++c) acc[c] += f0 * wt[c] + f1 * wt[16 + c];
    }
  }
  float mk = mask[p];
#pragma unroll
  for (int c = 0; c < 16; ++c) out[p * 16 + c] = (_Float16)(acc[c] * mk);
}

// -------------------------------------------- sparse maxpool 3x3 s2 (C=16)
static __global__ void k_maxpool(const _Float16* __restrict__ xin, const float* __restrict__ min_,
                                 float* __restrict__ xout, float* __restrict__ mout) {
  const int HW = 127 * 127;
  int p = blockIdx.x * blockDim.x + threadIdx.x;
  if (p >= 32 * HW) return;
  int b = p / HW;
  int r = p - b * HW;
  int y = r / 127, x = r - (r / 127) * 127;
  float mmax = 0.f;
  float best[16];
#pragma unroll
  for (int c = 0; c < 16; ++c) best[c] = -3.0e38f;
  for (int wy = 0; wy < 3; ++wy) {
    for (int wx = 0; wx < 3; ++wx) {
      int iy = 2 * y + wy, ix = 2 * x + wx;
      int ip = (b * 255 + iy) * 255 + ix;
      float mk = min_[ip];
      if (mk > mmax) mmax = mk;
      if (mk > 0.f) {
        const _Float16* px = xin + ip * 16;
#pragma unroll
        for (int c = 0; c < 16; ++c) {
          float v = (float)px[c];
          if (v > best[c]) best[c] = v;
        }
      }
    }
  }
  mout[p] = mmax;
#pragma unroll
  for (int c = 0; c < 16; ++c) xout[p * 16 + c] = (mmax > 0.f) ? best[c] : 0.f;
}

// ------------------------------------------------- mask pooling 3x3 VALID
static __global__ void k_poolmask(const float* __restrict__ min_, float* __restrict__ mout,
                                  int Hin, int Hout, int stride) {
  int P = 32 * Hout * Hout;
  int p = blockIdx.x * blockDim.x + threadIdx.x;
  if (p >= P) return;
  int HW = Hout * Hout;
  int b = p / HW;
  int r = p - b * HW;
  int y = r / Hout, x = r - (r / Hout) * Hout;
  float mm = 0.f;
  for (int wy = 0; wy < 3; ++wy)
    for (int wx = 0; wx < 3; ++wx) {
      int iy = y * stride + wy, ix = x * stride + wx;
      float v = min_[(b * Hin + iy) * Hin + ix];
      if (v > mm) mm = v;
    }
  mout[p] = mm;
}

// Swizzled-channel helper: slot s (0..31) within a 32-channel K-chunk maps to
// channel  cc*32 + (s>=16? 8:0) + (s&7) + ((s&15)>>3)*16  -- the exact order the
// WMMA A fragment wants per lane half (lo lanes: slots 0-15, hi lanes: 16-31).
__device__ __forceinline__ int swz_chan(int cc, int s) {
  return cc * 32 + ((s >> 4) << 3) + (s & 7) + (((s >> 3) & 1) << 4);
}

// ---- BN + ReLU + mask -> f16, zero-haloed padded + A-fragment-swizzled layout
// out is (32, H+2, H+2, CINP) halves, CINP = ceil(C/32)*32; halo & c>=C are 0.
static __global__ void k_bnrelu_pad(const float* __restrict__ x, const float* __restrict__ mask,
                                    const float* __restrict__ g, const float* __restrict__ be,
                                    const float* __restrict__ me, const float* __restrict__ va,
                                    _Float16* __restrict__ out, int H, int C, int CINP, int n) {
  int i = blockIdx.x * blockDim.x + threadIdx.x;
  if (i >= n) return;
  const int Hp = H + 2;
  int cs = i % CINP;
  int q = i / CINP;
  int xp = q % Hp;
  int q2 = q / Hp;
  int yp = q2 % Hp;
  int b = q2 / Hp;
  int c = swz_chan(cs >> 5, cs & 31);
  if (xp == 0 || xp > H || yp == 0 || yp > H || c >= C) {
    out[i] = (_Float16)0.f;
    return;
  }
  int p = (b * H + (yp - 1)) * H + (xp - 1);
  float y = (x[p * C + c] - me[c]) * rsqrtf(va[c] + EPS_BN) * g[c] + be[c];
  out[i] = (_Float16)(fmaxf(y, 0.f) * mask[p]);
}

// ---- f32 -> f16 cast, zero-haloed padded + A-fragment-swizzled layout
static __global__ void k_cast_pad(const float* __restrict__ x, _Float16* __restrict__ out,
                                  int H, int C, int CINP, int n) {
  int i = blockIdx.x * blockDim.x + threadIdx.x;
  if (i >= n) return;
  const int Hp = H + 2;
  int cs = i % CINP;
  int q = i / CINP;
  int xp = q % Hp;
  int q2 = q / Hp;
  int yp = q2 % Hp;
  int b = q2 / Hp;
  int c = swz_chan(cs >> 5, cs & 31);
  if (xp == 0 || xp > H || yp == 0 || yp > H || c >= C) {
    out[i] = (_Float16)0.f;
    return;
  }
  int p = (b * H + (yp - 1)) * H + (xp - 1);
  out[i] = (_Float16)x[p * C + c];
}

// -------------------------------------------- BN + ReLU + mask -> f16 (flat)
static __global__ void k_bnrelu(const float* __restrict__ x, const float* __restrict__ mask,
                                const float* __restrict__ g, const float* __restrict__ be,
                                const float* __restrict__ me, const float* __restrict__ va,
                                _Float16* __restrict__ out, int n, int C) {
  int i = blockIdx.x * blockDim.x + threadIdx.x;
  if (i >= n) return;
  int c = i % C;
  int p = i / C;
  float y = (x[i] - me[c]) * rsqrtf(va[c] + EPS_BN) * g[c] + be[c];
  y = fmaxf(y, 0.f) * mask[p];
  out[i] = (_Float16)y;
}

static __global__ void k_add(const float* __restrict__ a, const float* __restrict__ b,
                             float* __restrict__ d, int n) {
  int i = blockIdx.x * blockDim.x + threadIdx.x;
  if (i < n) d[i] = a[i] + b[i];
}

// -------------------------------- weight pre-pack into WMMA B-fragment layout
// dst[t][cc][nt][lane][j] = w[t][cc*32 + (lane>=16?8:0) + (j&7) + (j>=8?16:0)][nt*16 + lane%16]
// K zero-padded to 32*nch; N-tiles zero-padded to an even count (tilesNpad).
static __global__ void k_packw(const float* __restrict__ w, _Float16* __restrict__ dst,
                               int Cin, int Cout, int nch, int tilesNpad, int total) {
  int i = blockIdx.x * blockDim.x + threadIdx.x;
  if (i >= total) return;
  int j = i & 15;
  int lane = (i >> 4) & 31;
  int rest = i >> 9;
  int tilesN = Cout >> 4;
  int nt = rest % tilesNpad;
  int tc = rest / tilesNpad;
  int cc = tc % nch;
  int t = tc / nch;
  int k = (cc << 5) + ((lane >= 16) ? 8 : 0) + (j & 7) + ((j >> 3) << 4);
  int n = (nt << 4) + (lane & 15);
  float v = (k < Cin && nt < tilesN) ? w[(t * Cin + k) * Cout + n] : 0.f;
  dst[i] = (_Float16)v;
}

// --------------------------------------- implicit-GEMM 3x3 conv via WMMA
// Input: zero-haloed, A-fragment-swizzled f16 activations (32, Hp, Hp, CINP).
// Each A fragment is ONE aligned v16h load; B fragments come pre-packed.
// One wave = 32(M) x 32(N) tile: 2 A frags x 2 B frags -> 4 WMMAs per K-chunk.
template <int CIN, int COUT, int STRIDE, int PAD>
static __global__ __launch_bounds__(256)
void k_conv3x3_wmma(const _Float16* __restrict__ in, const _Float16* __restrict__ wpk,
                    const float* __restrict__ omask, float* __restrict__ out,
                    int Hp, int Hout) {
  constexpr int tilesN = COUT >> 4;
  constexpr int tilesNpad = (tilesN + 1) & ~1;
  constexpr int tilesP = tilesNpad >> 1;
  constexpr int NCH = (CIN + 31) >> 5;
  constexpr int CINP = NCH << 5;                 // swizzled halves per pixel
  constexpr int ccStride = tilesNpad << 9;       // weight halves per K-chunk step
  constexpr bool TWO = (COUT > 16);              // second N-subtile exists
  constexpr int OFF = 1 - PAD;                   // halo origin shift

  const int Wout = Hout;
  const int Mtot = 32 * Hout * Wout;             // % 32 == 0 (B = 32)
  const int tilesM = Mtot >> 5;                  // 32-row M tiles
  const int wtile = blockIdx.x * (blockDim.x >> 5) + (threadIdx.x >> 5);
  if (wtile >= tilesM * tilesP) return;
  const int tm = wtile / tilesP;
  const int pn = wtile - tm * tilesP;
  const bool has2 = TWO && ((pn * 32 + 32) <= COUT);

  const int lane = threadIdx.x & 31;
  const int row = lane & 15;
  const bool hi = lane >= 16;

  const int HWo = Hout * Wout;
  const int m0 = tm * 32 + row;
  const int m1 = m0 + 16;
  const int bb0 = m0 / HWo;
  const int q0 = m0 - bb0 * HWo;
  const int oy0 = q0 / Wout, ox0 = q0 - (q0 / Wout) * Wout;
  const int bb1 = m1 / HWo;
  const int q1 = m1 - bb1 * HWo;
  const int oy1 = q1 / Wout, ox1 = q1 - (q1 / Wout) * Wout;
  const int n = pn * 32 + row;

  // padded input coords: pixel (y,x) lives at (y+1, x+1)
  const int iyA = oy0 * STRIDE + OFF, ixA = ox0 * STRIDE + OFF;
  const int iyB = oy1 * STRIDE + OFF, ixB = ox1 * STRIDE + OFF;
  const int lsel = hi ? 16 : 0;                  // lane-half block in swizzle
  const _Float16* apA = in + ((bb0 * Hp + iyA) * Hp + ixA) * CINP + lsel;
  const _Float16* apB = in + ((bb1 * Hp + iyB) * Hp + ixB) * CINP + lsel;
  const _Float16* wbase = wpk + (pn << 10) + (lane << 4);

  v8f acc00 = {}, acc01 = {}, acc10 = {}, acc11 = {};

#pragma unroll
  for (int ky = 0; ky < 3; ++ky) {
#pragma unroll
    for (int kx = 0; kx < 3; ++kx) {
      const int tapo = (ky * Hp + kx) * CINP;    // wave-uniform immediate
      const _Float16* wt = wbase + (ky * 3 + kx) * (NCH * ccStride);
#pragma unroll
      for (int cc = 0; cc < NCH; ++cc) {
        v16h a0 = *(const v16h*)(apA + tapo + cc * 32);
        v16h a1 = *(const v16h*)(apB + tapo + cc * 32);
        const _Float16* wc = wt + cc * ccStride;
        v16h b0 = *(const v16h*)wc;
        acc00 = __builtin_amdgcn_wmma_f32_16x16x32_f16(false, a0, false, b0,
                                                       (short)0, acc00, false, false);
        acc10 = __builtin_amdgcn_wmma_f32_16x16x32_f16(false, a1, false, b0,
                                                       (short)0, acc10, false, false);
        if (TWO) {  // branchless: padded N-tile holds zero weights when !has2
          v16h b1 = *(const v16h*)(wc + 512);
          acc01 = __builtin_amdgcn_wmma_f32_16x16x32_f16(false, a0, false, b1,
                                                         (short)0, acc01, false, false);
          acc11 = __builtin_amdgcn_wmma_f32_16x16x32_f16(false, a1, false, b1,
                                                         (short)0, acc11, false, false);
        }
      }
    }
  }

  const int mb0 = tm * 32 + (hi ? 8 : 0);
  const int mb1 = mb0 + 16;
  const float* mp0 = omask + mb0;
  const float* mp1 = omask + mb1;
  float* op0 = out + mb0 * COUT + n;
  float* op1 = out + mb1 * COUT + n;
#pragma unroll
  for (int r = 0; r < 8; ++r) {
    op0[r * COUT] = acc00[r] * mp0[r];
    op1[r * COUT] = acc10[r] * mp1[r];
  }
  if (has2) {
#pragma unroll
    for (int r = 0; r < 8; ++r) {
      op0[r * COUT + 16] = acc01[r] * mp0[r];
      op1[r * COUT + 16] = acc11[r] * mp1[r];
    }
  }
}

template <int CIN, int COUT, int STRIDE, int PAD>
static void conv_launch(const _Float16* in, const _Float16* w, const float* om, float* out,
                        int Hin, int Hout, hipStream_t stream) {
  constexpr int tilesP = ((((COUT >> 4) + 1) & ~1) >> 1);
  int Mtot = 32 * Hout * Hout;
  long long tiles = (long long)(Mtot >> 5) * tilesP;
  unsigned blocks = (unsigned)((tiles + 7) / 8);  // 8 waves / block (wave32)
  k_conv3x3_wmma<CIN, COUT, STRIDE, PAD><<<blocks, 256, 0, stream>>>(in, w, om, out,
                                                                     Hin + 2, Hout);
}

// ------------------------------------------------------- final 256->10 linear
static __global__ void k_linear(const _Float16* __restrict__ h, const float* __restrict__ w,
                                const float* __restrict__ bias, float* __restrict__ out) {
  int t = blockIdx.x * blockDim.x + threadIdx.x;
  if (t >= 32 * 10) return;
  int b = t / 10, c = t - (t / 10) * 10;
  float s = bias[c];
  for (int k = 0; k < 256; ++k) s += (float)h[b * 256 + k] * w[k * 10 + c];
  out[t] = s;
}

// ============================================================== host driver
extern "C" void kernel_launch(void* const* d_in, const int* in_sizes, int n_in,
                              void* d_out, int out_size, void* d_ws, size_t ws_size,
                              hipStream_t stream) {
  (void)in_sizes; (void)n_in; (void)out_size; (void)ws_size;
  static const int STG[6][4] = {{16, 16, 2, 1}, {16, 32, 2, 2}, {32, 48, 2, 2},
                                {48, 96, 2, 2}, {96, 144, 2, 2}, {144, 192, 2, 2}};

  // ---- unpack inputs (jax pytree flatten order; None leaves dropped) ----
  int idx = 0;
  const float* feats  = (const float*)d_in[idx++];
  const float* w_stem = (const float*)d_in[idx++];
  const float* bnp[6][2][2][4];   // [stage][rep][bn1/bn2][g,b,m,v]
  const float* wconv[6][2][2];    // [stage][rep][w1/w2]
  const float* wscf[6];           // shortcut weight (rep 0, strided stages)
  for (int s = 0; s < 6; ++s) {
    wscf[s] = nullptr;
    for (int r = 0; r < 2; ++r) {
      for (int q = 0; q < 4; ++q) bnp[s][r][0][q] = (const float*)d_in[idx++];
      wconv[s][r][0] = (const float*)d_in[idx++];
      for (int q = 0; q < 4; ++q) bnp[s][r][1][q] = (const float*)d_in[idx++];
      wconv[s][r][1] = (const float*)d_in[idx++];
      if (r == 0 && STG[s][3] > 1) wscf[s] = (const float*)d_in[idx++];
    }
  }
  const float* bnf[4]; for (int q = 0; q < 4; ++q) bnf[q] = (const float*)d_in[idx++];
  const float* w_head = (const float*)d_in[idx++];
  const float* bnh[4]; for (int q = 0; q < 4; ++q) bnh[q] = (const float*)d_in[idx++];
  const float* w_lin = (const float*)d_in[idx++];
  const float* b_lin = (const float*)d_in[idx++];
  const int* coords  = (const int*)d_in[idx++];

  // ---- workspace bump allocator (deterministic per call) ----
  unsigned char* wp = (unsigned char*)d_ws;
  auto alloc = [&](size_t bytes) -> void* {
    void* r = (void*)wp;
    wp += (bytes + 255) & ~(size_t)255;
    return r;
  };
  auto nb = [](long long n) { return (unsigned)((n + 255) / 256); };
  auto cpad = [](int C) { return ((C + 31) >> 5) << 5; };

  const int P0 = 32 * 255 * 255;
  const int P1 = 32 * 127 * 127;
  const int P2 = 32 * 63 * 63;
  const size_t PPAD = (size_t)32 * 129 * 129 * 32;  // max padded+swizzled f16 elems

  float*    dense0 = (float*)alloc((size_t)P0 * 2 * sizeof(float));
  float*    mask0  = (float*)alloc((size_t)P0 * sizeof(float));
  _Float16* xstem  = (_Float16*)alloc((size_t)P0 * 16 * sizeof(_Float16));
  float*    xA     = (float*)alloc((size_t)P1 * 16 * sizeof(float));
  float*    xB     = (float*)alloc((size_t)P1 * 16 * sizeof(float));
  float*    mA     = (float*)alloc((size_t)P1 * sizeof(float));
  float*    mB     = (float*)alloc((size_t)P1 * sizeof(float));
  float*    hbuf   = (float*)alloc((size_t)P1 * 16 * sizeof(float));
  float*    h2buf  = (float*)alloc((size_t)P1 * 16 * sizeof(float));
  float*    scbuf  = (float*)alloc((size_t)P2 * 32 * sizeof(float));
  _Float16* h16    = (_Float16*)alloc(PPAD * sizeof(_Float16));
  _Float16* x16    = (_Float16*)alloc(PPAD * sizeof(_Float16));

  // ---- weights -> packed f16 fragment layout (N-tiles padded to even) ----
  auto packw = [&](const float* src, int Cin, int Cout) -> const _Float16* {
    int nch = (Cin + 31) >> 5;
    int tilesNpad = ((Cout >> 4) + 1) & ~1;
    int total = 9 * nch * tilesNpad * 512;
    _Float16* dst = (_Float16*)alloc((size_t)total * sizeof(_Float16));
    k_packw<<<nb(total), 256, 0, stream>>>(src, dst, Cin, Cout, nch, tilesNpad, total);
    return dst;
  };
  const _Float16* wc16[6][2][2];
  const _Float16* wsc16[6];
  for (int s = 0; s < 6; ++s) {
    int ci = STG[s][0], co = STG[s][1];
    for (int r = 0; r < 2; ++r) {
      int Ci = (r == 0) ? ci : co;
      wc16[s][r][0] = packw(wconv[s][r][0], Ci, co);
      wc16[s][r][1] = packw(wconv[s][r][1], co, co);
    }
    wsc16[s] = wscf[s] ? packw(wscf[s], ci, co) : nullptr;
  }
  const _Float16* whead16 = packw(w_head, 192, 256);

  // runtime -> template dispatch over the 12 layer configs
  auto run_conv = [&](const _Float16* in, const _Float16* w, const float* om, float* out,
                      int Hin, int Hout, int Cin, int Cout, int st, int pad) {
    if (st == 1 && pad == 1) {
      if      (Cin == 16  && Cout == 16)  conv_launch<16, 16, 1, 1>(in, w, om, out, Hin, Hout, stream);
      else if (Cin == 32  && Cout == 32)  conv_launch<32, 32, 1, 1>(in, w, om, out, Hin, Hout, stream);
      else if (Cin == 48  && Cout == 48)  conv_launch<48, 48, 1, 1>(in, w, om, out, Hin, Hout, stream);
      else if (Cin == 96  && Cout == 96)  conv_launch<96, 96, 1, 1>(in, w, om, out, Hin, Hout, stream);
      else if (Cin == 144 && Cout == 144) conv_launch<144, 144, 1, 1>(in, w, om, out, Hin, Hout, stream);
      else if (Cin == 192 && Cout == 192) conv_launch<192, 192, 1, 1>(in, w, om, out, Hin, Hout, stream);
    } else if (st == 2) {
      if      (Cin == 16  && Cout == 32)  conv_launch<16, 32, 2, 0>(in, w, om, out, Hin, Hout, stream);
      else if (Cin == 32  && Cout == 48)  conv_launch<32, 48, 2, 0>(in, w, om, out, Hin, Hout, stream);
      else if (Cin == 48  && Cout == 96)  conv_launch<48, 96, 2, 0>(in, w, om, out, Hin, Hout, stream);
      else if (Cin == 96  && Cout == 144) conv_launch<96, 144, 2, 0>(in, w, om, out, Hin, Hout, stream);
      else if (Cin == 144 && Cout == 192) conv_launch<144, 192, 2, 0>(in, w, om, out, Hin, Hout, stream);
    } else {  // st == 1, pad == 0: head conv
      conv_launch<192, 256, 1, 0>(in, w, om, out, Hin, Hout, stream);
    }
  };

  // ---- scatter -> dense grid + mask ----
  k_zero<<<nb(P0 * 2), 256, 0, stream>>>(dense0, P0 * 2);
  k_zero<<<nb(P0), 256, 0, stream>>>(mask0, P0);
  k_scatter<<<nb(200000), 256, 0, stream>>>(coords, feats, dense0, mask0, 200000);

  // ---- stem conv + sparse maxpool ----
  k_stem<<<nb(P0), 256, 0, stream>>>(dense0, mask0, w_stem, xstem);
  k_maxpool<<<nb(P1), 256, 0, stream>>>(xstem, mask0, xA, mA);

  // ---- residual stages ----
  float* x = xA;  float* xo = xB;
  float* mk = mA; float* mko = mB;
  int H = 127;
  for (int s = 0; s < 6; ++s) {
    const int cin = STG[s][0], cout = STG[s][1], stride = STG[s][3];
    for (int r = 0; r < 2; ++r) {
      const int Ci = (r == 0) ? cin : cout;
      const int st = (r == 0) ? stride : 1;
      const int Hin = H;
      const int Hout = (st == 1) ? Hin : (Hin - 3) / 2 + 1;
      const int pad = (st == 1) ? 1 : 0;
      const int Pout = 32 * Hout * Hout;
      const int nPadIn = 32 * (Hin + 2) * (Hin + 2) * cpad(Ci);
      const int nPadOut = 32 * (Hout + 2) * (Hout + 2) * cpad(cout);

      float* omask = mk;
      if (st != 1) {
        k_poolmask<<<nb(Pout), 256, 0, stream>>>(mk, mko, Hin, Hout, st);
        omask = mko;
      }
      // h = conv1(bn_relu(x)) * omask
      k_bnrelu_pad<<<nb(nPadIn), 256, 0, stream>>>(
          x, mk, bnp[s][r][0][0], bnp[s][r][0][1], bnp[s][r][0][2], bnp[s][r][0][3],
          h16, Hin, Ci, cpad(Ci), nPadIn);
      run_conv(h16, wc16[s][r][0], omask, hbuf, Hin, Hout, Ci, cout, st, pad);
      // h = conv2(bn_relu(h)) * omask
      k_bnrelu_pad<<<nb(nPadOut), 256, 0, stream>>>(
          hbuf, omask, bnp[s][r][1][0], bnp[s][r][1][1], bnp[s][r][1][2], bnp[s][r][1][3],
          h16, Hout, cout, cpad(cout), nPadOut);
      run_conv(h16, wc16[s][r][1], omask, h2buf, Hout, Hout, cout, cout, 1, 1);
      // shortcut
      const float* scp = x;
      if (r == 0 && wsc16[s]) {
        k_cast_pad<<<nb(nPadIn), 256, 0, stream>>>(x, x16, Hin, Ci, cpad(Ci), nPadIn);
        run_conv(x16, wsc16[s], omask, scbuf, Hin, Hout, Ci, cout, st, pad);
        scp = scbuf;
      }
      k_add<<<nb((long long)Pout * cout), 256, 0, stream>>>(h2buf, scp, xo, Pout * cout);
      // rotate buffers
      float* t = x; x = xo; xo = t;
      if (st != 1) { float* tm = mk; mk = mko; mko = tm; }
      H = Hout;
    }
  }

  // ---- head: bn_final -> 3x3 VALID conv (3->1) -> bn_head -> linear ----
  const int nPadHead = 32 * 5 * 5 * 192;
  k_bnrelu_pad<<<nb(nPadHead), 256, 0, stream>>>(x, mk, bnf[0], bnf[1], bnf[2], bnf[3],
                                                 h16, 3, 192, 192, nPadHead);
  k_poolmask<<<nb(32), 256, 0, stream>>>(mk, mko, 3, 1, 1);
  run_conv(h16, whead16, mko, hbuf, 3, 1, 192, 256, 1, 0);
  k_bnrelu<<<nb(32 * 256), 256, 0, stream>>>(hbuf, mko, bnh[0], bnh[1], bnh[2], bnh[3],
                                             h16, 32 * 256, 256);
  k_linear<<<2, 256, 0, stream>>>(h16, w_lin, b_lin, (float*)d_out);
}
